// CustomizedLSTM_84928683311425
// MI455X (gfx1250) — matrix operand order repaired and checked by reading, not compiled, not run on hardware
//
#include <hip/hip_runtime.h>
#include <hip/hip_bf16.h>
#include <math.h>
#include <stdint.h>

// ---------------------------------------------------------------------------
// MI455X (gfx1250) relu-LSTM stack.
//  - bf16 WMMA (v_wmma_f32_16x16x32_bf16) for all GEMMs, f32 accumulate
//  - Tensor Data Mover (tensor_load_to_lds + s_wait_tensorcnt) stages LDS
//    tiles; GEMM double-buffers so the DMA overlaps WMMA work
//  - recurrence: persistent scan kernel, grid barrier per timestep
// ---------------------------------------------------------------------------

typedef __bf16 bf16_t;
typedef __attribute__((ext_vector_type(16))) __bf16 v16bf;
typedef __attribute__((ext_vector_type(8)))  __bf16 v8bf;
typedef __attribute__((ext_vector_type(8)))  float  v8f;
typedef __attribute__((ext_vector_type(4)))  unsigned int u32x4;
typedef __attribute__((ext_vector_type(8)))  int i32x8;
typedef __attribute__((ext_vector_type(4)))  int i32x4;

#define BATCH 128
#define SEQLEN 512
#define BT (BATCH * SEQLEN)

__device__ __forceinline__ float sigmoidf_(float x) {
  return 1.0f / (1.0f + __expf(-x));
}

// Low 32 bits of a generic pointer to __shared__ = LDS byte address
// (flat aperture).
__device__ __forceinline__ unsigned lds_offset(const void* p) {
  return (unsigned)(uintptr_t)p;
}

// TDM 2D tile load: nrows rows of rowlen_elts bf16 elements, global row
// stride row_stride_elts, packed contiguously into LDS at lds_addr.
// D# group0: count=1 | lds_addr | global_addr | type=2 (ISA 8.3)
// D# group1: data_size=2B, tensor_dim0/1, tile_dim0/1, dim0 stride (ISA 8.4)
__device__ __forceinline__ void tdm_load_2d(unsigned lds_addr, const void* gptr,
                                            unsigned rowlen_elts, unsigned nrows,
                                            unsigned row_stride_elts) {
  unsigned long long ga = (unsigned long long)(uintptr_t)gptr;
  u32x4 g0;
  g0[0] = 1u;                                                  // count=1
  g0[1] = lds_addr;                                            // [63:32]
  g0[2] = (unsigned)ga;                                        // [95:64]
  g0[3] = (unsigned)((ga >> 32) & 0x01FFFFFFu) | (2u << 30);   // ga hi | type=2
  i32x8 g1;
  g1[0] = (int)(1u << 16);                                     // data_size = 2B
  g1[1] = (int)((rowlen_elts & 0xFFFFu) << 16);                // tensor_dim0 lo16
  g1[2] = (int)(((rowlen_elts >> 16) & 0xFFFFu) |
                ((nrows & 0xFFFFu) << 16));                    // dim0 hi | dim1 lo
  g1[3] = (int)(((nrows >> 16) & 0xFFFFu) |
                ((rowlen_elts & 0xFFFFu) << 16));              // dim1 hi | tile_dim0
  g1[4] = (int)(nrows & 0xFFFFu);                              // tile_dim1, tile_dim2=0
  g1[5] = (int)row_stride_elts;                                // tensor_dim0_stride lo32
  g1[6] = 0;
  g1[7] = 0;
  i32x4 z4 = {0, 0, 0, 0};
#if defined(__clang_major__) && __clang_major__ >= 23
  i32x8 z8 = {0, 0, 0, 0, 0, 0, 0, 0};
  __builtin_amdgcn_tensor_load_to_lds(g0, g1, z4, z4, z8, 0);
#else
  __builtin_amdgcn_tensor_load_to_lds(g0, g1, z4, z4, 0);
#endif
}

// A-matrix fragment, 16x32 bf16 (ISA 7.12.2): lane L -> row M = L%16,
// elements 0..7 = K[kbase..kbase+7], 8..15 = K[kbase+16..kbase+23], kbase=(L>=16)*8
__device__ __forceinline__ v16bf load_frag_a(const bf16_t* base, int row0, int stride, int kb) {
  int lane = threadIdx.x & 31;
  const bf16_t* p = base + (size_t)(row0 + (lane & 15)) * stride + kb + ((lane >> 4) << 3);
  v8bf lo = *(const v8bf*)p;
  v8bf hi = *(const v8bf*)(p + 16);
  v16bf r;
#pragma unroll
  for (int e = 0; e < 8; ++e) { r[e] = lo[e]; r[e + 8] = hi[e]; }
  return r;
}

// B-matrix fragment, 32x16 bf16: lane L -> col N = L%16,
// elements 0..7 = K[kbase..kbase+7], 8..15 = K[kbase+16..kbase+23], kbase=(L>=16)*8. `base` = transposed weight.
__device__ __forceinline__ v16bf load_frag_b(const bf16_t* base, int row0, int stride, int kb) {
  int lane = threadIdx.x & 31;
  const bf16_t* p = base + (size_t)(row0 + (lane & 15)) * stride + kb + ((lane >> 4) << 3);
  v8bf lo = *(const v8bf*)p;
  v8bf hi = *(const v8bf*)(p + 16);
  v16bf r;
#pragma unroll
  for (int e = 0; e < 8; ++e) { r[e] = lo[e]; r[e + 8] = hi[e]; }
  return r;
}

// ---------------------- embedding gather -> padded bf16 --------------------
__global__ __launch_bounds__(256) void k_embed(const int* __restrict__ tokens,
                                               const float* __restrict__ emb,
                                               bf16_t* __restrict__ out,
                                               int kpad, int K, int total) {
  int idx = blockIdx.x * blockDim.x + threadIdx.x;
  if (idx >= total) return;
  int row = idx / kpad;
  int k = idx - row * kpad;
  float v = 0.0f;
  if (k < K) v = emb[(size_t)tokens[row] * K + k];
  out[idx] = (bf16_t)v;
}

// -------------- weight convert+transpose: WT[n][kpad] = W[k][n] ------------
__global__ __launch_bounds__(256) void k_wconv(const float* __restrict__ W,
                                               bf16_t* __restrict__ WT,
                                               int K, int N, int KPAD) {
  int idx = blockIdx.x * blockDim.x + threadIdx.x;
  if (idx >= N * KPAD) return;
  int n = idx / KPAD;
  int k = idx - n * KPAD;
  WT[idx] = (bf16_t)((k < K) ? W[(size_t)k * N + n] : 0.0f);
}

// -------- batched projection GEMM: C[M,N] = A[M,KPAD] @ WT[N,KPAD]^T + b ----
// Block = 4 waves, 128x64 output tile; wave w does rows w*32..w*32+31 (two A
// fragments share every B fragment). B tile (64 x 32K bf16) is TDM-loaded
// into a double-buffered LDS tile so the DMA overlaps WMMA on the other tile.
__global__ __launch_bounds__(128) void k_gemm_xw(const bf16_t* __restrict__ A,
                                                 const bf16_t* __restrict__ WT,
                                                 const float* __restrict__ bias,
                                                 float* __restrict__ C,
                                                 int KPAD, int N) {
  __shared__ __align__(16) bf16_t tb[2][64 * 32];
  const int tid = threadIdx.x;
  const int wave = tid >> 5, lane = tid & 31;
  const int n0 = blockIdx.x * 64;
  const int m0 = blockIdx.y * 128;
  const int mw = m0 + wave * 32;

  v8f acc[2][4];
#pragma unroll
  for (int h = 0; h < 2; ++h)
#pragma unroll
    for (int j = 0; j < 4; ++j)
#pragma unroll
      for (int e = 0; e < 8; ++e) acc[h][j][e] = 0.0f;

  const int niter = KPAD >> 5;
  if (tid < 32)  // wave 0 drives the TDM pipeline (EXEC is ignored by TDM)
    tdm_load_2d(lds_offset(&tb[0][0]), WT + (size_t)n0 * KPAD, 32u, 64u, (unsigned)KPAD);

  for (int i = 0; i < niter; ++i) {
    const int kb = i << 5;
    if (tid < 32) __builtin_amdgcn_s_wait_tensorcnt(0);   // tile i landed
    __syncthreads();                                      // publish; tb[(i+1)&1] free
    if (tid < 32 && (i + 1) < niter)
      tdm_load_2d(lds_offset(&tb[(i + 1) & 1][0]),
                  WT + (size_t)n0 * KPAD + kb + 32, 32u, 64u, (unsigned)KPAD);

    v16bf a0 = load_frag_a(A, mw, KPAD, kb);
    v16bf a1 = load_frag_a(A, mw + 16, KPAD, kb);
    const bf16_t* tbi = &tb[i & 1][0];
#pragma unroll
    for (int j = 0; j < 4; ++j) {
      v16bf b = load_frag_b(tbi, j * 16, 32, 0);
      acc[0][j] = __builtin_amdgcn_wmma_f32_16x16x32_bf16(false, a0, false, b,
                                                          (short)0, acc[0][j], false, false);
      acc[1][j] = __builtin_amdgcn_wmma_f32_16x16x32_bf16(false, a1, false, b,
                                                          (short)0, acc[1][j], false, false);
    }
  }

#pragma unroll
  for (int h = 0; h < 2; ++h)
#pragma unroll
    for (int j = 0; j < 4; ++j) {
      int col = n0 + j * 16 + (lane & 15);
      float bv = bias[col];
#pragma unroll
      for (int e = 0; e < 8; ++e) {
        int row = mw + h * 16 + ((lane < 16) ? e : e + 8);
        C[(size_t)row * N + col] = acc[h][j][e] + bv;
      }
    }
}

// ------------------------------ LSTM scan ----------------------------------
// Grid: (u/64) x (BATCH/16). 4 waves/block, wave g computes gate g's
// [16 x 64] pre-activation tile via WMMA over K=u. h tile is TDM-staged into
// LDS each step; c-state lives in LDS; grid barrier separates timesteps.
__global__ __launch_bounds__(128) void k_lstm_scan(const float* __restrict__ xg,   // [BT,4u]
                                                   const bf16_t* __restrict__ UT,  // [4u,u]
                                                   bf16_t* __restrict__ hseq,      // [BT,u]
                                                   bf16_t* __restrict__ hcur0,
                                                   bf16_t* __restrict__ hcur1,
                                                   unsigned* __restrict__ bar_cnt,
                                                   unsigned* __restrict__ bar_gen,
                                                   int u, int nwg) {
  __shared__ __align__(16) bf16_t hbuf[16 * 512];
  __shared__ float gbuf[4][16][64];
  __shared__ float cbuf[16 * 64];

  const int tid = threadIdx.x;
  const int wave = tid >> 5, lane = tid & 31;
  const int n0 = blockIdx.x * 64;
  const int m0 = blockIdx.y * 16;
  const int G4u = 4 * u;
  const int gatebase = wave * u + n0;   // columns of gate `wave` in the 4u axis

#pragma unroll
  for (int q = 0; q < 8; ++q) cbuf[tid * 8 + q] = 0.0f;
  __syncthreads();

  for (int t = 0; t < SEQLEN; ++t) {
    const bf16_t* hprev = (t & 1) ? hcur1 : hcur0;
    bf16_t* hnext = (t & 1) ? hcur0 : hcur1;

    // TDM: stage h_{t-1} rows m0..m0+15 (full u columns) into LDS
    if (tid < 32) {
      tdm_load_2d(lds_offset(hbuf), hprev + (size_t)m0 * u, (unsigned)u, 16u, (unsigned)u);
      __builtin_amdgcn_s_wait_tensorcnt(0);
    }
    __syncthreads();

    // accumulator starts at xg (bias + x@W already folded in)
    v8f acc[4];
#pragma unroll
    for (int j = 0; j < 4; ++j) {
      int col = gatebase + j * 16 + (lane & 15);
#pragma unroll
      for (int e = 0; e < 8; ++e) {
        int m = (lane < 16) ? e : e + 8;
        acc[j][e] = xg[((size_t)(m0 + m) * SEQLEN + t) * G4u + col];
      }
    }

    for (int kb = 0; kb < u; kb += 32) {
      v16bf a = load_frag_a(hbuf, 0, u, kb);        // h tile from LDS
#pragma unroll
      for (int j = 0; j < 4; ++j) {
        v16bf b = load_frag_b(UT, gatebase + j * 16, u, kb);  // U slice (L2-resident)
        acc[j] = __builtin_amdgcn_wmma_f32_16x16x32_bf16(false, a, false, b,
                                                         (short)0, acc[j], false, false);
      }
    }

    // exchange gate tiles through LDS for the elementwise cell update
#pragma unroll
    for (int j = 0; j < 4; ++j) {
      int n = j * 16 + (lane & 15);
#pragma unroll
      for (int e = 0; e < 8; ++e) {
        int m = (lane < 16) ? e : e + 8;
        gbuf[wave][m][n] = acc[j][e];
      }
    }
    __syncthreads();

#pragma unroll
    for (int q = 0; q < 8; ++q) {
      int idx = tid * 8 + q;
      int m = idx >> 6, n = idx & 63;
      float gi = sigmoidf_(gbuf[0][m][n]);
      float gf = sigmoidf_(gbuf[1][m][n]);
      float gc = fmaxf(gbuf[2][m][n], 0.0f);
      float go = sigmoidf_(gbuf[3][m][n]);
      float cc = gf * cbuf[idx] + gi * gc;
      cbuf[idx] = cc;
      float h = go * fmaxf(cc, 0.0f);
      bf16_t hb = (bf16_t)h;
      hnext[(size_t)(m0 + m) * u + (n0 + n)] = hb;
      hseq[((size_t)(m0 + m) * SEQLEN + t) * u + (n0 + n)] = hb;
    }

    // grid barrier (generation counter; safe across launches & graph replays)
    __syncthreads();
    if (tid == 0) {
      __threadfence();
      unsigned g = __hip_atomic_load(bar_gen, __ATOMIC_ACQUIRE, __HIP_MEMORY_SCOPE_AGENT);
      unsigned p = __hip_atomic_fetch_add(bar_cnt, 1u, __ATOMIC_ACQ_REL, __HIP_MEMORY_SCOPE_AGENT);
      if (p == (unsigned)nwg - 1u) {
        __hip_atomic_store(bar_cnt, 0u, __ATOMIC_RELAXED, __HIP_MEMORY_SCOPE_AGENT);
        __hip_atomic_fetch_add(bar_gen, 1u, __ATOMIC_ACQ_REL, __HIP_MEMORY_SCOPE_AGENT);
      } else {
        while (__hip_atomic_load(bar_gen, __ATOMIC_ACQUIRE, __HIP_MEMORY_SCOPE_AGENT) == g) {
          __builtin_amdgcn_s_sleep(2);
        }
      }
    }
    __syncthreads();
  }
}

// --------------------------- dense head ------------------------------------
__global__ __launch_bounds__(128) void k_head(const bf16_t* __restrict__ hseq, // [BT,256]
                                              const float* __restrict__ Wd,
                                              const float* __restrict__ bd,
                                              const float* __restrict__ Wc,
                                              const float* __restrict__ bc,
                                              float* __restrict__ out) {
  int b = threadIdx.x;
  const bf16_t* h = hseq + ((size_t)b * SEQLEN + (SEQLEN - 1)) * 256;
  float z = bc[0];
  for (int d = 0; d < 64; ++d) {
    float s = bd[d];
    for (int j = 0; j < 256; ++j) s += (float)h[j] * Wd[j * 64 + d];
    z += fmaxf(s, 0.0f) * Wc[d];
  }
  out[b] = 1.0f / (1.0f + __expf(-z));
}

// ---------------------------------------------------------------------------
extern "C" void kernel_launch(void* const* d_in, const int* in_sizes, int n_in,
                              void* d_out, int out_size, void* d_ws, size_t ws_size,
                              hipStream_t stream) {
  (void)in_sizes; (void)n_in; (void)out_size; (void)ws_size;
  const int*   tokens = (const int*)d_in[0];
  const float* emb = (const float*)d_in[1];
  const float* W0 = (const float*)d_in[2];
  const float* U0 = (const float*)d_in[3];
  const float* b0 = (const float*)d_in[4];
  const float* W1 = (const float*)d_in[5];
  const float* U1 = (const float*)d_in[6];
  const float* b1 = (const float*)d_in[7];
  const float* W2 = (const float*)d_in[8];
  const float* U2 = (const float*)d_in[9];
  const float* b2 = (const float*)d_in[10];
  const float* Wd = (const float*)d_in[11];
  const float* bd = (const float*)d_in[12];
  const float* Wc = (const float*)d_in[13];
  const float* bc = (const float*)d_in[14];

  char* ws = (char*)d_ws;
  size_t off = 0;
  auto carve = [&](size_t bytes) -> char* {
    char* p = ws + off;
    off += (bytes + 255) & ~(size_t)255;
    return p;
  };
  unsigned* bar   = (unsigned*)carve(256);
  bf16_t*   hcur0 = (bf16_t*)carve((size_t)BATCH * 512 * 2);
  bf16_t*   hcur1 = (bf16_t*)carve((size_t)BATCH * 512 * 2);
  bf16_t*   bufA  = (bf16_t*)carve((size_t)BT * 512 * 2);
  bf16_t*   bufB  = (bf16_t*)carve((size_t)BT * 512 * 2);
  float*    xg    = (float*)carve((size_t)BT * 2048 * 4);
  bf16_t*   W0T   = (bf16_t*)carve((size_t)1024 * 320 * 2);
  bf16_t*   U0T   = (bf16_t*)carve((size_t)1024 * 256 * 2);
  bf16_t*   W1T   = (bf16_t*)carve((size_t)2048 * 256 * 2);
  bf16_t*   U1T   = (bf16_t*)carve((size_t)2048 * 512 * 2);
  bf16_t*   W2T   = (bf16_t*)carve((size_t)1024 * 512 * 2);
  bf16_t*   U2T   = (bf16_t*)carve((size_t)1024 * 256 * 2);

  hipMemsetAsync(bar, 0, 256, stream);

  auto wconv = [&](const float* W, bf16_t* WT, int K, int N, int KPAD) {
    int total = N * KPAD;
    k_wconv<<<(total + 255) / 256, 256, 0, stream>>>(W, WT, K, N, KPAD);
  };
  wconv(W0, W0T, 300, 1024, 320);
  wconv(U0, U0T, 256, 1024, 256);
  wconv(W1, W1T, 256, 2048, 256);
  wconv(U1, U1T, 512, 2048, 512);
  wconv(W2, W2T, 512, 1024, 512);
  wconv(U2, U2T, 256, 1024, 256);

  {
    int total = BT * 320;
    k_embed<<<(total + 255) / 256, 256, 0, stream>>>(tokens, emb, bufA, 320, 300, total);
  }

  // ---- layer 0: D=320(pad of 300), u=256 ----
  k_gemm_xw<<<dim3(1024 / 64, BT / 128), 128, 0, stream>>>(bufA, W0T, b0, xg, 320, 1024);
  hipMemsetAsync(hcur0, 0, (size_t)BATCH * 512 * 2 * 2, stream);  // h0 = 0 (both parities)
  {
    dim3 g(256 / 64, BATCH / 16);
    k_lstm_scan<<<g, 128, 0, stream>>>(xg, U0T, bufB, hcur0, hcur1, bar, bar + 1, 256, g.x * g.y);
  }
  // ---- layer 1: D=256, u=512 ----
  k_gemm_xw<<<dim3(2048 / 64, BT / 128), 128, 0, stream>>>(bufB, W1T, b1, xg, 256, 2048);
  hipMemsetAsync(hcur0, 0, (size_t)BATCH * 512 * 2 * 2, stream);
  {
    dim3 g(512 / 64, BATCH / 16);
    k_lstm_scan<<<g, 128, 0, stream>>>(xg, U1T, bufA, hcur0, hcur1, bar, bar + 1, 512, g.x * g.y);
  }
  // ---- layer 2: D=512, u=256 ----
  k_gemm_xw<<<dim3(1024 / 64, BT / 128), 128, 0, stream>>>(bufA, W2T, b2, xg, 512, 1024);
  hipMemsetAsync(hcur0, 0, (size_t)BATCH * 512 * 2 * 2, stream);
  {
    dim3 g(256 / 64, BATCH / 16);
    k_lstm_scan<<<g, 128, 0, stream>>>(xg, U2T, bufB, hcur0, hcur1, bar, bar + 1, 256, g.x * g.y);
  }
  k_head<<<1, 128, 0, stream>>>(bufB, Wd, bd, Wc, bc, (float*)d_out);
}
